// DifuscoGNNLayer_46746424050133
// MI455X (gfx1250) — matrix operand order repaired and checked by reading, not compiled, not run on hardware
//
#include <hip/hip_runtime.h>
#include <stdint.h>

// ---------------------------------------------------------------------------
// DifuscoGNNLayer for MI455X (gfx1250, wave32, WMMA)
// Dominant GEMM (262144x256 @ 256x256) done with split-bf16 WMMA
// (x = bf16_hi + bf16_lo; A*B ~= Ah*Bh + Ah*Bl + Al*Bh), W3 resident in LDS.
// ---------------------------------------------------------------------------

typedef __attribute__((ext_vector_type(16))) __bf16 bf16x16;
typedef __attribute__((ext_vector_type(8)))  float  f32x8;

#define NTOK 512
#define DD   256
#define NROW (NTOK * NTOK)          // 262144 flattened (i,j) rows
#define ROWS_PER_BLK 32
#define NBLK (NROW / ROWS_PER_BLK)  // 8192

__device__ __forceinline__ unsigned short f2bf(float f) {
    union { float f; unsigned u; } v; v.f = f;
    unsigned u = v.u;
    unsigned r = u + 0x7FFFu + ((u >> 16) & 1u);   // round-to-nearest-even
    return (unsigned short)(r >> 16);
}
__device__ __forceinline__ float bf2f(unsigned short b) {
    union { unsigned u; float f; } v; v.u = ((unsigned)b) << 16;
    return v.f;
}

// ---------------- prep: 4 transposed fp32 weight copies -------------------
__global__ void k_transpose4(const float* __restrict__ W1, const float* __restrict__ W2,
                             const float* __restrict__ W4, const float* __restrict__ W5,
                             float* __restrict__ T1, float* __restrict__ T2,
                             float* __restrict__ T4, float* __restrict__ T5) {
    const float* src; float* dst;
    switch (blockIdx.y) {
        case 0:  src = W1; dst = T1; break;
        case 1:  src = W2; dst = T2; break;
        case 2:  src = W4; dst = T4; break;
        default: src = W5; dst = T5; break;
    }
    int k = blockIdx.x, d = threadIdx.x;
    dst[d * DD + k] = src[k * DD + d];
}

// ---------------- prep: t_bias = t_emb @ W_t^T ----------------------------
__global__ void k_tbias(const float* __restrict__ temb, const float* __restrict__ Wt,
                        float* __restrict__ tb) {
    int k = threadIdx.x;
    float a = 0.f;
    for (int d = 0; d < DD; ++d) a += temb[d] * Wt[k * DD + d];
    tb[k] = a;
}

// ---------------- prep: W3 -> swizzled split bf16 (hi/lo) -----------------
// element (n,d) stored at n*256 + ((d>>3 + n)&31)*8 + (d&7): 16B chunks are
// rotated by n per row so 16-lane ds_load_b128 fragment reads are bank-free.
__global__ void k_w3split(const float* __restrict__ W3,
                          unsigned short* __restrict__ w3h,
                          unsigned short* __restrict__ w3l) {
    int n = blockIdx.x, d = threadIdx.x;
    float x = W3[n * DD + d];
    unsigned short ah = f2bf(x);
    unsigned short al = f2bf(x - bf2f(ah));
    int cp  = (((d >> 3) + n) & 31);
    int off = n * DD + cp * 8 + (d & 7);
    w3h[off] = ah; w3l[off] = al;
}

// ---------------- prep: hi_t = h@W1^T + t_bias ;  hj = h@W2^T -------------
__global__ void k_hi_hj(const float* __restrict__ h, const float* __restrict__ W1T,
                        const float* __restrict__ W2T, const float* __restrict__ tb,
                        float* __restrict__ hi_t, float* __restrict__ hj) {
    __shared__ float sH[8][DD];
    int i0 = blockIdx.x * 8, t = threadIdx.x;
    for (int r = 0; r < 8; ++r) sH[r][t] = h[(i0 + r) * DD + t];
    __syncthreads();
    float a1[8] = {}, a2[8] = {};
    for (int d = 0; d < DD; ++d) {
        float w1 = W1T[d * DD + t], w2 = W2T[d * DD + t];
#pragma unroll
        for (int r = 0; r < 8; ++r) { float hv = sH[r][d]; a1[r] += hv * w1; a2[r] += hv * w2; }
    }
    float tbv = tb[t];
    for (int r = 0; r < 8; ++r) {
        hi_t[(i0 + r) * DD + t] = a1[r] + tbv;
        hj  [(i0 + r) * DD + t] = a2[r];
    }
}

// ---------------- main: WMMA GEMM + bias + LN + ReLU + gate logits --------
__global__ __launch_bounds__(256, 1) void k_edge_main(
    const float* __restrict__ e, const float* __restrict__ hi_t, const float* __restrict__ hj,
    const unsigned short* __restrict__ w3h, const unsigned short* __restrict__ w3l,
    const float* __restrict__ ln_w, const float* __restrict__ ln_b,
    const float* __restrict__ gate_w, const float* __restrict__ gate_b,
    float* __restrict__ eout, float* __restrict__ logits) {

    __shared__ __align__(16) unsigned short sBh[65536];   // W3 hi (LDS-resident)
    __shared__ __align__(16) unsigned short sBl[65536];   // W3 lo
    __shared__ __align__(16) unsigned short sAh[8192];    // 32x256 A hi tile
    __shared__ __align__(16) unsigned short sAl[8192];    // 32x256 A lo tile
    __shared__ float sHi[256];                            // hi_t row for this i
    __shared__ float sS[128], sQ[128], sL[128];           // LN / logit partials

    const int tid  = threadIdx.x;
    const int w    = tid >> 5;         // wave 0..7
    const int lane = tid & 31;
    const int nn   = lane & 15;
    const int s    = lane >> 4;
    const int mt   = w & 1;            // which 16-row tile of the 32-row block
    const int cg   = w >> 1;           // 64-column group 0..3

    // Fill LDS-resident W3 splits (256KB) once per persistent workgroup.
    {
        const uint4* gh = (const uint4*)w3h;
        const uint4* gl = (const uint4*)w3l;
        uint4* shh = (uint4*)sBh;
        uint4* shl = (uint4*)sBl;
        for (int idx = tid; idx < 8192; idx += 256) { shh[idx] = gh[idx]; shl[idx] = gl[idx]; }
    }
    __syncthreads();

    const float gb = gate_b[0];

    for (int T = blockIdx.x; T < NBLK; T += gridDim.x) {
        const int R0 = T * ROWS_PER_BLK;
        const int i  = R0 >> 9;
        const int j0 = R0 & 511;

        // ---- stage A: 32 rows of e, split to bf16 hi/lo, chunk-swizzled ----
#pragma unroll
        for (int it = 0; it < 8; ++it) {
            int f   = it * 256 + tid;          // float4 index 0..2047
            int row = f >> 6;
            int d4  = f & 63;
            float4 x = ((const float4*)(e + (size_t)(R0 + row) * DD))[d4];
            int cp  = ((d4 >> 1) + row) & 31;
            int off = row * 256 + cp * 8 + (d4 & 1) * 4;
            unsigned short a0 = f2bf(x.x), a1 = f2bf(x.y), a2 = f2bf(x.z), a3 = f2bf(x.w);
            sAh[off + 0] = a0; sAh[off + 1] = a1; sAh[off + 2] = a2; sAh[off + 3] = a3;
            sAl[off + 0] = f2bf(x.x - bf2f(a0));
            sAl[off + 1] = f2bf(x.y - bf2f(a1));
            sAl[off + 2] = f2bf(x.z - bf2f(a2));
            sAl[off + 3] = f2bf(x.w - bf2f(a3));
        }
        sHi[tid] = hi_t[i * DD + tid];
        __syncthreads();

        // ---- GEMM: 16 rows x 64 cols per wave, K=256 in 8 steps of 32 ----
        f32x8 acc[4] = {};
        const int arow = mt * 16 + nn;     // A-matrix row for this lane
#pragma unroll
        for (int kk = 0; kk < 8; ++kk) {
            union { uint4 q[2]; bf16x16 v; } aH, aL;
            int c0  = kk * 4 + s;
            int c1  = kk * 4 + 2 + s;
            int cp0 = (c0 + arow) & 31;
            int cp1 = (c1 + arow) & 31;
            aH.q[0] = *(const uint4*)((const char*)sAh + arow * 512 + cp0 * 16);
            aH.q[1] = *(const uint4*)((const char*)sAh + arow * 512 + cp1 * 16);
            aL.q[0] = *(const uint4*)((const char*)sAl + arow * 512 + cp0 * 16);
            aL.q[1] = *(const uint4*)((const char*)sAl + arow * 512 + cp1 * 16);
#pragma unroll
            for (int t = 0; t < 4; ++t) {
                int n   = cg * 64 + t * 16 + nn;
                int bc0 = kk * 4 + s;
                int bp0 = (bc0 + n) & 31;
                int bp1 = (bc0 + 2 + n) & 31;
                union { uint4 q[2]; bf16x16 v; } bH, bL;
                bH.q[0] = *(const uint4*)((const char*)sBh + n * 512 + bp0 * 16);
                bH.q[1] = *(const uint4*)((const char*)sBh + n * 512 + bp1 * 16);
                bL.q[0] = *(const uint4*)((const char*)sBl + n * 512 + bp0 * 16);
                bL.q[1] = *(const uint4*)((const char*)sBl + n * 512 + bp1 * 16);
                acc[t] = __builtin_amdgcn_wmma_f32_16x16x32_bf16(false, aH.v, false, bH.v,
                                                                 (short)0, acc[t], false, false);
                acc[t] = __builtin_amdgcn_wmma_f32_16x16x32_bf16(false, aH.v, false, bL.v,
                                                                 (short)0, acc[t], false, false);
                acc[t] = __builtin_amdgcn_wmma_f32_16x16x32_bf16(false, aL.v, false, bH.v,
                                                                 (short)0, acc[t], false, false);
            }
        }

        // ---- epilogue on fragments: +hi +hj +t_bias, LN partial sums ----
        float lw[4], lb[4], gw[4], hv[4];
        int cols[4];
#pragma unroll
        for (int t = 0; t < 4; ++t) {
            int col = cg * 64 + t * 16 + nn;
            cols[t] = col;
            hv[t] = sHi[col];
            lw[t] = ln_w[col];
            lb[t] = ln_b[col];
            gw[t] = gate_w[col];
        }
#pragma unroll
        for (int r = 0; r < 8; ++r) {
            int rowblk = mt * 16 + s * 8 + r;
            int j      = j0 + rowblk;
            float sum = 0.f, sq = 0.f;
#pragma unroll
            for (int t = 0; t < 4; ++t) {
                float v = acc[t][r] + hv[t] + hj[(size_t)j * DD + cols[t]];
                acc[t][r] = v;
                sum += v; sq += v * v;
            }
#pragma unroll
            for (int m = 1; m < 16; m <<= 1) {
                sum += __shfl_xor(sum, m, 32);
                sq  += __shfl_xor(sq,  m, 32);
            }
            if (nn == 0) { sS[cg * 32 + rowblk] = sum; sQ[cg * 32 + rowblk] = sq; }
        }
        __syncthreads();

        // ---- finalize LN + ReLU, store e_new, gate-logit partials ----
#pragma unroll
        for (int r = 0; r < 8; ++r) {
            int rowblk = mt * 16 + s * 8 + r;
            float S = sS[rowblk] + sS[32 + rowblk] + sS[64 + rowblk] + sS[96 + rowblk];
            float Q = sQ[rowblk] + sQ[32 + rowblk] + sQ[64 + rowblk] + sQ[96 + rowblk];
            float mean = S * (1.f / 256.f);
            float var  = Q * (1.f / 256.f) - mean * mean;
            float rstd = rsqrtf(var + 1e-5f);
            float lp = 0.f;
            size_t obase = (size_t)(R0 + rowblk) * DD;
#pragma unroll
            for (int t = 0; t < 4; ++t) {
                float v = (acc[t][r] - mean) * rstd * lw[t] + lb[t];
                v = fmaxf(v, 0.f);
                eout[obase + cols[t]] = v;
                lp += v * gw[t];
            }
#pragma unroll
            for (int m = 1; m < 16; m <<= 1) lp += __shfl_xor(lp, m, 32);
            if (nn == 0) sL[cg * 32 + rowblk] = lp;
        }
        __syncthreads();
        if (tid < 32) {
            logits[R0 + tid] = sL[tid] + sL[32 + tid] + sL[64 + tid] + sL[96 + tid] + gb;
        }
    }
}

// ---------------- softmax over j + weighted reduction of e_new ------------
__global__ void k_softagg(const float* __restrict__ logits, const float* __restrict__ e_new,
                          float* __restrict__ agg) {
    __shared__ float red[256];
    __shared__ float eta[512];
    int i = blockIdx.x, t = threadIdx.x;
    const float* lr = logits + i * 512;
    float a = lr[t], b = lr[t + 256];
    red[t] = fmaxf(a, b);
    __syncthreads();
    for (int o = 128; o > 0; o >>= 1) { if (t < o) red[t] = fmaxf(red[t], red[t + o]); __syncthreads(); }
    float mx = red[0];
    __syncthreads();
    float e0 = expf(a - mx), e1 = expf(b - mx);
    eta[t] = e0; eta[t + 256] = e1;
    red[t] = e0 + e1;
    __syncthreads();
    for (int o = 128; o > 0; o >>= 1) { if (t < o) red[t] += red[t + o]; __syncthreads(); }
    float inv = 1.f / red[0];
    __syncthreads();
    eta[t] *= inv; eta[t + 256] *= inv;
    __syncthreads();
    const float* eb = e_new + (size_t)i * 512 * DD;
    float acc = 0.f;
    for (int j = 0; j < 512; ++j) acc += eta[j] * eb[(size_t)j * DD + t];
    agg[i * DD + t] = acc;
}

// ---------------- final: h_new = h + relu(LN(h@W4^T + agg@W5^T)) ----------
__global__ void k_hupdate(const float* __restrict__ h, const float* __restrict__ agg,
                          const float* __restrict__ W4T, const float* __restrict__ W5T,
                          const float* __restrict__ lnw, const float* __restrict__ lnb,
                          float* __restrict__ out) {
    __shared__ float sH[8][DD];
    __shared__ float sA[8][DD];
    __shared__ float redS[256], redQ[256];
    int i0 = blockIdx.x * 8, t = threadIdx.x;
    for (int r = 0; r < 8; ++r) {
        sH[r][t] = h[(i0 + r) * DD + t];
        sA[r][t] = agg[(i0 + r) * DD + t];
    }
    __syncthreads();
    float acc[8] = {};
    for (int d = 0; d < DD; ++d) {
        float w4 = W4T[d * DD + t], w5 = W5T[d * DD + t];
#pragma unroll
        for (int r = 0; r < 8; ++r) acc[r] += sH[r][d] * w4 + sA[r][d] * w5;
    }
    float lw = lnw[t], lb = lnb[t];
    for (int r = 0; r < 8; ++r) {
        redS[t] = acc[r]; redQ[t] = acc[r] * acc[r];
        __syncthreads();
        for (int o = 128; o > 0; o >>= 1) {
            if (t < o) { redS[t] += redS[t + o]; redQ[t] += redQ[t + o]; }
            __syncthreads();
        }
        float mean = redS[0] * (1.f / 256.f);
        float var  = redQ[0] * (1.f / 256.f) - mean * mean;
        float rstd = rsqrtf(var + 1e-5f);
        __syncthreads();
        float v = (acc[r] - mean) * rstd * lw + lb;
        out[(i0 + r) * DD + t] = sH[r][t] + fmaxf(v, 0.f);
    }
}

// ---------------------------------------------------------------------------
extern "C" void kernel_launch(void* const* d_in, const int* in_sizes, int n_in,
                              void* d_out, int out_size, void* d_ws, size_t ws_size,
                              hipStream_t stream) {
    const float* h    = (const float*)d_in[0];
    const float* e    = (const float*)d_in[1];
    const float* temb = (const float*)d_in[2];
    const float* W1   = (const float*)d_in[3];
    const float* W2   = (const float*)d_in[4];
    const float* W3   = (const float*)d_in[5];
    const float* Wt   = (const float*)d_in[6];
    const float* W4   = (const float*)d_in[7];
    const float* W5   = (const float*)d_in[8];
    const float* gw   = (const float*)d_in[9];
    const float* gb   = (const float*)d_in[10];
    const float* lnew = (const float*)d_in[11];
    const float* lneb = (const float*)d_in[12];
    const float* lnhw = (const float*)d_in[13];
    const float* lnhb = (const float*)d_in[14];

    // workspace layout (~3.6 MB total)
    float* wsf    = (float*)d_ws;
    float* W1T    = wsf;               // 65536
    float* W2T    = wsf + 65536;       // 65536
    float* W4T    = wsf + 131072;      // 65536
    float* W5T    = wsf + 196608;      // 65536
    float* tb     = wsf + 262144;      // 256
    float* hi_t   = wsf + 262400;      // 131072
    float* hj     = wsf + 393472;      // 131072
    float* agg    = wsf + 524544;      // 65536
    float* logits = wsf + 590080;      // 262144
    unsigned short* w3h = (unsigned short*)((char*)d_ws + 3408896); // 65536 bf16
    unsigned short* w3l = w3h + 65536;                              // 65536 bf16

    float* h_new = (float*)d_out;            // 512*256
    float* e_new = (float*)d_out + 131072;   // 512*512*256

    k_transpose4<<<dim3(256, 4), 256, 0, stream>>>(W1, W2, W4, W5, W1T, W2T, W4T, W5T);
    k_tbias<<<1, 256, 0, stream>>>(temb, Wt, tb);
    k_w3split<<<256, 256, 0, stream>>>(W3, w3h, w3l);
    k_hi_hj<<<64, 256, 0, stream>>>(h, W1T, W2T, tb, hi_t, hj);
    k_edge_main<<<1024, 256, 0, stream>>>(e, hi_t, hj, w3h, w3l,
                                          lnew, lneb, gw, gb, e_new, logits);
    k_softagg<<<512, 256, 0, stream>>>(logits, e_new, agg);
    k_hupdate<<<64, 256, 0, stream>>>(h, agg, W4T, W5T, lnhw, lnhb, h_new);
}
